// LocationSlayerWhorl_6408091205837
// MI455X (gfx1250) — matrix operand order repaired and checked by reading, not compiled, not run on hardware
//
#include <hip/hip_runtime.h>

// ---------------------------------------------------------------------------
// SLAYER SRM forward on MI455X (gfx1250, wave32, WMMA).
// f16 WMMA GEMM (register-blocked 32x64 per wave, 8 wmma per K-step)
//   -> 100-tap causal FIR (LDS) -> sequential threshold/refractory (LDS ring).
// ---------------------------------------------------------------------------

typedef __attribute__((ext_vector_type(16))) _Float16 v16h;
typedef __attribute__((ext_vector_type(8)))  _Float16 v8h;
typedef __attribute__((ext_vector_type(8)))  float    v8f;

#define THETA 10.0f

// ---------------------------------------------------------------------------
// Constants: srm[k] = (k/10) * exp(1 - k/10), k=0..99
//            ref[k] = -2*10*(k+1) * exp(1 - (k+1)),  k=0..31
// ---------------------------------------------------------------------------
__global__ void init_consts_kernel(float* __restrict__ srm, float* __restrict__ refk) {
  int i = threadIdx.x;
  if (i < 100) {
    float t = (float)i;
    srm[i] = (t * 0.1f) * __expf(1.0f - t * 0.1f);
  }
  if (i < 32) {
    float t = (float)(i + 1);
    refk[i] = -2.0f * THETA * t * __expf(1.0f - t);
  }
}

// f32 [Mv][Kv] -> f16 [Mp][Kp], zero padded
__global__ __launch_bounds__(256)
void pack_w_kernel(const float* __restrict__ w, _Float16* __restrict__ out,
                   int Mv, int Kv, int Mp, int Kp) {
  int idx = blockIdx.x * 256 + threadIdx.x;
  if (idx >= Mp * Kp) return;
  int k = idx % Kp, m = idx / Kp;
  float v = (m < Mv && k < Kv) ? w[(size_t)m * Kv + k] : 0.0f;
  out[idx] = (_Float16)v;
}

// x f32 [64][156][500] -> f16 [64][512][160]  (out[n][t][c], c contiguous)
__global__ __launch_bounds__(256)
void pack_x_tc_kernel(const float* __restrict__ x, _Float16* __restrict__ out) {
  int idx = blockIdx.x * 256 + threadIdx.x;
  if (idx >= 64 * 512 * 160) return;
  int c = idx % 160;
  int t = (idx / 160) % 512;
  int n = idx / (160 * 512);
  float v = (c < 156 && t < 500) ? x[((size_t)n * 156 + c) * 500 + t] : 0.0f;
  out[idx] = (_Float16)v;
}

// x f32 [64][156][500] -> f16 [64][192][512]  (out[n][j][t] = x[n][perm[j]][t])
__global__ __launch_bounds__(256)
void pack_x_perm_kernel(const float* __restrict__ x, const int* __restrict__ perm,
                        _Float16* __restrict__ out) {
  int idx = blockIdx.x * 256 + threadIdx.x;
  if (idx >= 64 * 192 * 512) return;
  int t = idx % 512;
  int j = (idx / 512) % 192;
  int n = idx / (512 * 192);
  float v = 0.0f;
  if (j < 156 && t < 500) v = x[((size_t)n * 156 + perm[j]) * 500 + t];
  out[idx] = (_Float16)v;
}

// ---------------------------------------------------------------------------
// WMMA GEMM, register-blocked:  Out[b][m][col] = sum_k A[m][k] * B[b][col][k]
//   A:   f16 [M][Kpad] row-major (K contiguous); M mult of 32, Kpad mult of 32
//   B:   f16 [batch][Npad][Kpad], K contiguous per column; Npad mult of 64
//   Out: f32 [batch][M][Nstride]
// Each wave computes a 32(M) x 64(N) macro-tile: 2 A-frags x 4 B-frags
// -> 8 v_wmma_f32_16x16x32_f16 per K-step (A and B loads amortized 4x / 2x).
// 4 waves per block along N-groups.  Control flow is wave-uniform (EXEC full).
// A/B use contiguous k = 16*hi + i (same order for both); f32 C/D layout per CDNA5 ISA 7.12.2.
// ---------------------------------------------------------------------------
__global__ __launch_bounds__(128)
void wmma_gemm_f16(const _Float16* __restrict__ A, const _Float16* __restrict__ B,
                   float* __restrict__ Out,
                   int M, int Kpad, int Npad, int Nstride, int NtileGroups) {
  const int lane = threadIdx.x & 31;
  const int wave = threadIdx.x >> 5;
  const int ng = blockIdx.x * 4 + wave;      // N tile-group (64 columns)
  if (ng >= NtileGroups) return;             // wave-uniform early-out
  const int mg = blockIdx.y;                 // M tile-group (32 rows)
  const int b  = blockIdx.z;

  const int lo = lane & 15;
  const int hi = lane >> 4;

  // A: lane holds row; halves i -> K = k0 + 16*hi + i
  const _Float16* Arow0 = A + (size_t)(mg * 32 + lo) * (size_t)Kpad + hi * 16;
  const _Float16* Arow1 = Arow0 + (size_t)16 * Kpad;
  // B: lane holds column; halves i -> K = k0 + 16*hi + i
  const _Float16* Bbase = B + ((size_t)b * Npad + (size_t)(ng * 64 + lo)) * (size_t)Kpad
                            + hi * 16;

  v8f acc[2][4] = {};
  for (int k0 = 0; k0 < Kpad; k0 += 32) {
    v16h af0, af1;
    {
      v8h a0 = *(const v8h*)(Arow0 + k0);
      v8h a1 = *(const v8h*)(Arow0 + k0 + 8);
#pragma unroll
      for (int i = 0; i < 8; ++i) { af0[i] = a0[i]; af0[8 + i] = a1[i]; }
    }
    {
      v8h a0 = *(const v8h*)(Arow1 + k0);
      v8h a1 = *(const v8h*)(Arow1 + k0 + 8);
#pragma unroll
      for (int i = 0; i < 8; ++i) { af1[i] = a0[i]; af1[8 + i] = a1[i]; }
    }
    v16h bf[4];
#pragma unroll
    for (int j = 0; j < 4; ++j)
      bf[j] = *(const v16h*)(Bbase + (size_t)j * 16 * Kpad + k0);   // 32B load
#pragma unroll
    for (int j = 0; j < 4; ++j) {
      acc[0][j] = __builtin_amdgcn_wmma_f32_16x16x32_f16(
          false, af0, false, bf[j], (short)0, acc[0][j], false, false);
      acc[1][j] = __builtin_amdgcn_wmma_f32_16x16x32_f16(
          false, af1, false, bf[j], (short)0, acc[1][j], false, false);
    }
  }

  // D layout: VGPR r, lanes0-15 -> M=r, lanes16-31 -> M=8+r; N = lo
#pragma unroll
  for (int mi = 0; mi < 2; ++mi) {
#pragma unroll
    for (int j = 0; j < 4; ++j) {
      float* Op = Out + ((size_t)b * M + (size_t)(mg * 32 + mi * 16 + hi * 8)) * (size_t)Nstride
                      + (ng * 64 + j * 16 + lo);
#pragma unroll
      for (int r = 0; r < 8; ++r) Op[(size_t)r * Nstride] = acc[mi][j][r];
    }
  }
}

// ---------------------------------------------------------------------------
// Causal FIR:  out[row][i] = sum_{k=0..min(i,99)} srm[k] * in[row][i-k]
// ---------------------------------------------------------------------------
__global__ __launch_bounds__(128)
void psp_conv_kernel(const float* __restrict__ in, float* __restrict__ out,
                     const float* __restrict__ srm, int L, int Lstride) {
  __shared__ float sk[100];
  __shared__ float srow[512];
  const int row = blockIdx.x;
  const float* ip = in + (size_t)row * Lstride;
  float* op = out + (size_t)row * Lstride;
  for (int i = threadIdx.x; i < 100; i += 128) sk[i] = srm[i];
  for (int i = threadIdx.x; i < L; i += 128) srow[i] = ip[i];
  __syncthreads();
  for (int i = threadIdx.x; i < L; i += 128) {
    float acc = 0.0f;
    const int kmax = (i + 1 < 100) ? (i + 1) : 100;
    for (int k = 0; k < kmax; ++k) acc = fmaf(sk[k], srow[i - k], acc);
    op[i] = acc;
  }
}

// ---------------------------------------------------------------------------
// Spike dynamics (hidden layers): one lane per (n,h) sequence; 32-slot
// refractory ring in LDS; spikes stored transposed f16 [n][t][h] (coalesced,
// K-contiguous B for the next GEMM).  Zero-fills pad rows t in [L, LpadOut).
// ---------------------------------------------------------------------------
__global__ __launch_bounds__(256)
void spike_dyn_hidden_kernel(const float* __restrict__ u, const float* __restrict__ refk,
                             _Float16* __restrict__ sout,
                             int Hdim, int L, int Lstride, int LpadOut, int outTStride,
                             int rowsTotal) {
  __shared__ float ring[256][33];   // stride 33 -> conflict-free
  __shared__ float sref[32];
  const int tid = threadIdx.x;
  if (tid < 32) sref[tid] = refk[tid];
#pragma unroll
  for (int k = 0; k < 32; ++k) ring[tid][k] = 0.0f;
  __syncthreads();

  const int r = blockIdx.x * 256 + tid;
  const bool active = (r < rowsTotal);
  const int rr = active ? r : 0;
  const int n = rr / Hdim, h = rr % Hdim;
  const float* up = u + (size_t)rr * Lstride;
  _Float16* sp = sout + (size_t)n * LpadOut * outTStride + h;

  for (int t = 0; t < L; ++t) {
    const int slot = t & 31;
    float v = up[t] + ring[tid][slot];
    ring[tid][slot] = 0.0f;                       // becomes slot for t+32
    float s = (v >= THETA) ? 1.0f : 0.0f;
    if (s != 0.0f) {
#pragma unroll
      for (int k = 1; k <= 32; ++k) ring[tid][(t + k) & 31] += sref[k - 1];
    }
    if (active) sp[(size_t)t * outTStride] = (_Float16)s;
  }
  for (int t = L; t < LpadOut; ++t)
    if (active) sp[(size_t)t * outTStride] = (_Float16)0.0f;
}

// Final-layer dynamics: writes f32 straight into the output tensor slice.
__global__ __launch_bounds__(256)
void spike_dyn_final_kernel(const float* __restrict__ u, const float* __restrict__ refk,
                            float* __restrict__ out,
                            int Ovalid, int Mstride, int L, int Lstride,
                            int outStride, int outOff, int rowsTotal) {
  __shared__ float ring[256][33];
  __shared__ float sref[32];
  const int tid = threadIdx.x;
  if (tid < 32) sref[tid] = refk[tid];
#pragma unroll
  for (int k = 0; k < 32; ++k) ring[tid][k] = 0.0f;
  __syncthreads();

  const int r = blockIdx.x * 256 + tid;
  const bool active = (r < rowsTotal);
  const int rr = active ? r : 0;
  const int n = rr / Ovalid, o = rr % Ovalid;
  const float* up = u + ((size_t)n * Mstride + o) * Lstride;
  float* op = out + (size_t)(n * Ovalid + o) * outStride + outOff;

  for (int t = 0; t < L; ++t) {
    const int slot = t & 31;
    float v = up[t] + ring[tid][slot];
    ring[tid][slot] = 0.0f;
    float s = (v >= THETA) ? 1.0f : 0.0f;
    if (s != 0.0f) {
#pragma unroll
      for (int k = 1; k <= 32; ++k) ring[tid][(t + k) & 31] += sref[k - 1];
    }
    if (active) op[t] = s;
  }
}

// ---------------------------------------------------------------------------
// Host-side orchestration (graph-capture safe: only kernel launches on stream)
// ---------------------------------------------------------------------------
extern "C" void kernel_launch(void* const* d_in, const int* in_sizes, int n_in,
                              void* d_out, int out_size, void* d_ws, size_t ws_size,
                              hipStream_t stream) {
  (void)in_sizes; (void)n_in; (void)out_size; (void)ws_size;
  const float* x     = (const float*)d_in[0];  // (64,156,1,1,500)
  const float* w_fc1 = (const float*)d_in[1];  // (512,156)
  const float* w_fc2 = (const float*)d_in[2];  // (20,512)
  const float* w_l1  = (const float*)d_in[3];  // (512,500)
  const float* w_l2  = (const float*)d_in[4];  // (20,512)
  const int*   perm  = (const int*)d_in[5];    // (156,)
  float* out = (float*)d_out;                  // (64,20,656)

  char* p = (char*)d_ws;
  auto alloc = [&](size_t bytes) -> char* {
    char* r = p; p += (bytes + 255) & ~(size_t)255; return r;
  };
  float*    srm  = (float*)alloc(100 * 4);
  float*    refk = (float*)alloc(32 * 4);
  _Float16* w1h  = (_Float16*)alloc((size_t)512 * 160 * 2);      // [512][160]
  _Float16* w2h  = (_Float16*)alloc((size_t)32 * 512 * 2);       // [32][512]
  _Float16* wl1h = (_Float16*)alloc((size_t)512 * 512 * 2);      // [512][512]
  _Float16* wl2h = (_Float16*)alloc((size_t)32 * 512 * 2);       // [32][512]
  _Float16* x_tc = (_Float16*)alloc((size_t)64 * 512 * 160 * 2); // [n][t][c]
  _Float16* x_pm = (_Float16*)alloc((size_t)64 * 192 * 512 * 2); // [n][taxel][t]
  _Float16* s1h  = (_Float16*)alloc((size_t)64 * 512 * 512 * 2); // [n][t][h]
  _Float16* l1h  = (_Float16*)alloc((size_t)64 * 192 * 512 * 2); // [n][taxel][h]
  float*    bufA = (float*)alloc((size_t)64 * 512 * 512 * 4);    // GEMM out
  float*    bufB = (float*)alloc((size_t)64 * 512 * 512 * 4);    // conv out

  init_consts_kernel<<<1, 128, 0, stream>>>(srm, refk);
  pack_w_kernel<<<(512 * 160 + 255) / 256, 256, 0, stream>>>(w_fc1, w1h, 512, 156, 512, 160);
  pack_w_kernel<<<(32 * 512 + 255) / 256, 256, 0, stream>>>(w_fc2, w2h, 20, 512, 32, 512);
  pack_w_kernel<<<(512 * 512 + 255) / 256, 256, 0, stream>>>(w_l1, wl1h, 512, 500, 512, 512);
  pack_w_kernel<<<(32 * 512 + 255) / 256, 256, 0, stream>>>(w_l2, wl2h, 20, 512, 32, 512);
  pack_x_tc_kernel<<<(64 * 512 * 160 + 255) / 256, 256, 0, stream>>>(x, x_tc);
  pack_x_perm_kernel<<<(64 * 192 * 512 + 255) / 256, 256, 0, stream>>>(x, perm, x_pm);

  // ---- temporal path, layer 1: w_fc1 @ x -> (64,512,500) ----
  { dim3 g(2, 16, 64);   // Ngroups=8, Mgroups=16
    wmma_gemm_f16<<<g, 128, 0, stream>>>(w1h, x_tc, bufA, 512, 160, 512, 512, 8); }
  psp_conv_kernel<<<64 * 512, 128, 0, stream>>>(bufA, bufB, srm, 500, 512);
  spike_dyn_hidden_kernel<<<(64 * 512 + 255) / 256, 256, 0, stream>>>(
      bufB, refk, s1h, 512, 500, 512, 512, 512, 64 * 512);

  // ---- temporal path, layer 2: w_fc2 @ s1 -> (64,20,500) -> d_out[:, :, 0:500] ----
  { dim3 g(2, 1, 64);    // Ngroups=8, Mgroups=1
    wmma_gemm_f16<<<g, 128, 0, stream>>>(w2h, s1h, bufA, 32, 512, 512, 512, 8); }
  psp_conv_kernel<<<64 * 32, 128, 0, stream>>>(bufA, bufB, srm, 500, 512);
  spike_dyn_final_kernel<<<(64 * 20 + 255) / 256, 256, 0, stream>>>(
      bufB, refk, out, 20, 32, 500, 512, 656, 0, 64 * 20);

  // ---- location path, layer 1: w_loc1 @ x_perm^T -> (64,512,156) ----
  { dim3 g(1, 16, 64);   // Ngroups=3 (192 cols), Mgroups=16
    wmma_gemm_f16<<<g, 128, 0, stream>>>(wl1h, x_pm, bufA, 512, 512, 192, 192, 3); }
  psp_conv_kernel<<<64 * 512, 128, 0, stream>>>(bufA, bufB, srm, 156, 192);
  spike_dyn_hidden_kernel<<<(64 * 512 + 255) / 256, 256, 0, stream>>>(
      bufB, refk, l1h, 512, 156, 192, 192, 512, 64 * 512);

  // ---- location path, layer 2: w_loc2 @ l1 -> (64,20,156) -> d_out[:, :, 500:656] ----
  { dim3 g(1, 1, 64);    // Ngroups=3, Mgroups=1
    wmma_gemm_f16<<<g, 128, 0, stream>>>(wl2h, l1h, bufA, 32, 512, 192, 192, 3); }
  psp_conv_kernel<<<64 * 32, 128, 0, stream>>>(bufA, bufB, srm, 156, 192);
  spike_dyn_final_kernel<<<(64 * 20 + 255) / 256, 256, 0, stream>>>(
      bufB, refk, out, 20, 32, 156, 192, 656, 500, 64 * 20);
}
